// SlidingWindowAttention_74552042324780
// MI455X (gfx1250) — matrix operand order repaired and checked by reading, not compiled, not run on hardware
//
#include <hip/hip_runtime.h>
#include <hip/hip_bf16.h>
#include <math.h>

// ---------------------------------------------------------------------------
// SlidingWindowAttention (LN -> stride4 -> QKV+RoPE -> causal attn -> proj ->
// upsample + residual) for gfx1250. All GEMM/attention math runs on
// v_wmma_f32_16x16x32_f16; weight/K/V panels are staged into LDS with the
// Tensor Data Mover, software-pipelined (issue next DMA, s_wait_tensorcnt 1).
// Workspace layout (needs >= 64 MB):
//   [ 0,  8M) xs      f16 [4096][1024]   normalized, strided tokens
//   [ 8M,14M) WqkvT   f16 [3072][1024]   W_qkv transposed (B-operand layout)
//   [14M,16M) WoutT   f16 [1024][1024]   W_out transposed
//   [16M,24M) q       f16 [B*H][Ts][64]  RoPE'd
//   [24M,32M) k       f16 [B*H][Ts][64]  RoPE'd
//   [32M,40M) vT      f16 [B*H][64][Ts]  transposed for P@V B-operand
//   [40M,48M) attnout f16 [4096][1024]
//   [48M,64M) proj    f32 [4096][1024]
// ---------------------------------------------------------------------------

typedef __attribute__((ext_vector_type(16))) _Float16     v16h;
typedef __attribute__((ext_vector_type(8)))  _Float16     v8h;
typedef __attribute__((ext_vector_type(8)))  float        v8f;
typedef __attribute__((ext_vector_type(4)))  unsigned int u32x4;
typedef __attribute__((ext_vector_type(8)))  int          i32x8;
typedef __attribute__((ext_vector_type(4)))  int          i32x4;

#define TS   1024
#define DD   1024
#define NH   16
#define HD   64
#define TF   4096

#if defined(__has_builtin)
#if __has_builtin(__builtin_amdgcn_tensor_load_to_lds) && \
    __has_builtin(__builtin_amdgcn_s_wait_tensorcnt)
#define HAVE_TDM 1
#endif
#endif

__device__ __forceinline__ v16h make_v16(v8h lo, v8h hi) {
  v16h r;
#pragma unroll
  for (int i = 0; i < 8; ++i) { r[i] = lo[i]; r[i + 8] = hi[i]; }
  return r;
}

// A-operand (16x32 f16): lane row = lane&15, K = hf*8+e (e<8), 16+hf*8+(e-8)
__device__ __forceinline__ v16h ldA(const _Float16* row, int k0, int hf) {
  v8h lo = *(const v8h*)(row + k0 + hf * 8);
  v8h hi = *(const v8h*)(row + k0 + 16 + hf * 8);
  return make_v16(lo, hi);
}
// B-operand (32x16 f16): lane col = lane&15, K = hf*8+e (e<8), 16+hf*8+(e-8)
__device__ __forceinline__ v16h ldB(const _Float16* row, int k0, int hf) {
  v8h lo = *(const v8h*)(row + k0 + hf * 8);
  v8h hi = *(const v8h*)(row + k0 + 16 + hf * 8);
  return make_v16(lo, hi);
}

__device__ __forceinline__ v8f wmma16(v16h a, v16h b, v8f c) {
  return __builtin_amdgcn_wmma_f32_16x16x32_f16(false, a, false, b, (short)0, c,
                                                false, false);
}

#ifdef HAVE_TDM
// TDM: DMA a (tile1 rows x tile0*8 bytes) tile, row stride stride0*8 bytes,
// from global into contiguous LDS. tile1 == 0 -> 1-D contiguous transfer.
__device__ __forceinline__ void tdm_load(const void* gsrc, void* lds_dst,
                                         int tile0, int tile1, int stride0) {
  unsigned lds_addr = (unsigned)(uintptr_t)lds_dst;  // low 32 bits = LDS offset
  unsigned long long ga = (unsigned long long)(uintptr_t)gsrc;
  u32x4 g0;
  g0[0] = 1u;                                     // count=1, user descriptor
  g0[1] = lds_addr;                               // lds_addr (bytes)
  g0[2] = (unsigned)(ga & 0xFFFFFFFFu);           // global_addr[31:0]
  g0[3] = (unsigned)((ga >> 32) & 0x01FFFFFFu)    // global_addr[56:32]
          | (2u << 30);                           // type = 2 ("image")
  i32x8 g1;
  g1[0] = (3 << 16);            // workgroup_mask=0, data_size=3 (8 bytes)
  g1[1] = 0;                    // abar=0, tensor_dim0[15:0]=0  (td0 = 1<<20)
  g1[2] = 16;                   // tensor_dim0[31:16]=16, tensor_dim1[15:0]=0
  g1[3] = 16 | (tile0 << 16);   // tensor_dim1[31:16]=16 (td1=1<<20), tile_dim0
  g1[4] = tile1 & 0xFFFF;       // tile_dim1, tile_dim2=0
  g1[5] = stride0;              // tensor_dim0_stride (8B units)
  g1[6] = 0;
  g1[7] = 0;
  i32x4 z4 = {0, 0, 0, 0};
#if __clang_major__ >= 23
  i32x8 z8 = {0, 0, 0, 0, 0, 0, 0, 0};
  __builtin_amdgcn_tensor_load_to_lds(g0, g1, z4, z4, z8, 0);
#else
  __builtin_amdgcn_tensor_load_to_lds(g0, g1, z4, z4, 0);
#endif
}
#endif

// --------------------------- LayerNorm + stride4 ---------------------------
__global__ void k_ln(const float* __restrict__ x, const float* __restrict__ w,
                     const float* __restrict__ bn, _Float16* __restrict__ xs) {
  int row = blockIdx.x;                    // 0 .. B*Ts-1
  int b = row >> 10, ts = row & 1023;
  const float* xp = x + ((size_t)b * TF + (size_t)ts * 4) * DD;
  int tid = threadIdx.x;                   // 256 threads, 4 floats each
  float4 v = ((const float4*)xp)[tid];
  float s  = v.x + v.y + v.z + v.w;
  float s2 = v.x * v.x + v.y * v.y + v.z * v.z + v.w * v.w;
#pragma unroll
  for (int o = 1; o < 32; o <<= 1) {
    s += __shfl_xor(s, o, 32);
    s2 += __shfl_xor(s2, o, 32);
  }
  __shared__ float sh[16];
  int wv = tid >> 5, ln = tid & 31;
  if (ln == 0) { sh[wv] = s; sh[8 + wv] = s2; }
  __syncthreads();
  float t1 = 0.f, t2 = 0.f;
#pragma unroll
  for (int i = 0; i < 8; ++i) { t1 += sh[i]; t2 += sh[8 + i]; }
  float mu  = t1 * (1.0f / DD);
  float var = t2 * (1.0f / DD) - mu * mu;
  float inv = rsqrtf(var + 1e-5f);
  float4 wv4 = ((const float4*)w)[tid];
  float4 bv4 = ((const float4*)bn)[tid];
  _Float16* op = xs + (size_t)row * DD + tid * 4;
  op[0] = (_Float16)((v.x - mu) * inv * wv4.x + bv4.x);
  op[1] = (_Float16)((v.y - mu) * inv * wv4.y + bv4.y);
  op[2] = (_Float16)((v.z - mu) * inv * wv4.z + bv4.z);
  op[3] = (_Float16)((v.w - mu) * inv * wv4.w + bv4.w);
}

// --------------------------- weight transposes ---------------------------
__global__ void k_t_qkv(const float* __restrict__ W, _Float16* __restrict__ WT) {
  size_t i = (size_t)blockIdx.x * blockDim.x + threadIdx.x;   // 3072*1024
  if (i >= (size_t)3072 * 1024) return;
  size_t c = i >> 10, k = i & 1023;
  WT[i] = (_Float16)W[k * 3072 + c];
}
__global__ void k_t_out(const float* __restrict__ W, _Float16* __restrict__ WT) {
  size_t i = (size_t)blockIdx.x * blockDim.x + threadIdx.x;   // 1024*1024
  if (i >= (size_t)1024 * 1024) return;
  size_t c = i >> 10, k = i & 1023;
  WT[i] = (_Float16)W[k * 1024 + c];
}

// --------------------------- QKV GEMM + RoPE scatter ---------------------------
// 8 waves (8 M-tiles) share one head-slot's 64-row weight panel, TDM-staged
// into LDS in 128-deep K chunks, double-buffered. Each wave: 16x64 output;
// RoPE rotate-half partners (d, d+32) both live in this wave.
__global__ void k_qkv(const _Float16* __restrict__ xs,
                      const _Float16* __restrict__ WT,
                      const float* __restrict__ bias,
                      _Float16* __restrict__ qb, _Float16* __restrict__ kb,
                      _Float16* __restrict__ vT) {
  __shared__ alignas(16) _Float16 panel[2][64 * 128];   // 2 x 16 KB
  int wv = threadIdx.x >> 5;
  int lane = threadIdx.x & 31;
  int hf = lane >> 4, ln = lane & 15;
  int mblock = blockIdx.x / 48;            // 32 blocks of 8 M-tiles
  int hh = blockIdx.x - mblock * 48;       // head slot: 0..47 (3*16)
  int rowbase = (mblock * 8 + wv) * 16;
  int c0 = hh * 64;
  const _Float16* arow = xs + (size_t)(rowbase + ln) * DD;
  const _Float16* wsrc = WT + ((size_t)c0 << 10);
  v8f a0 = {}, a1 = {}, a2 = {}, a3 = {};
#ifdef HAVE_TDM
  if (wv == 0) tdm_load(wsrc, panel[0], 32, 64, 256);
#endif
  for (int kc = 0; kc < DD; kc += 128) {
    int cur = (kc >> 7) & 1;
#ifdef HAVE_TDM
    if (wv == 0) {
      if (kc + 128 < DD) {
        tdm_load(wsrc + kc + 128, panel[cur ^ 1], 32, 64, 256);
        __builtin_amdgcn_s_wait_tensorcnt(1);   // current panel done
      } else {
        __builtin_amdgcn_s_wait_tensorcnt(0);
      }
    }
#else
    {  // cooperative fallback: 256 threads x 32 f16
      int t = threadIdx.x;
      const _Float16* src = wsrc + ((size_t)(t >> 2) << 10) + kc + (t & 3) * 32;
      _Float16* dst = panel[cur] + (t >> 2) * 128 + (t & 3) * 32;
#pragma unroll
      for (int u = 0; u < 4; ++u)
        *(v8h*)(dst + u * 8) = *(const v8h*)(src + u * 8);
    }
#endif
    __syncthreads();
    const _Float16* pb = panel[cur];
#pragma unroll
    for (int k0 = 0; k0 < 128; k0 += 32) {
      __builtin_prefetch(arow + kc + k0 + 256, 0, 1);   // global_prefetch_b8
      v16h a = ldA(arow, kc + k0, hf);
      a0 = wmma16(a, ldB(pb + (size_t)(ln)      * 128, k0, hf), a0);
      a1 = wmma16(a, ldB(pb + (size_t)(16 + ln) * 128, k0, hf), a1);
      a2 = wmma16(a, ldB(pb + (size_t)(32 + ln) * 128, k0, hf), a2);
      a3 = wmma16(a, ldB(pb + (size_t)(48 + ln) * 128, k0, hf), a3);
    }
    __syncthreads();
  }
  int which = hh >> 4, h = hh & 15;
  int b = rowbase >> 10;
  int tsbase = rowbase & 1023;
  float bs0 = bias[c0 + ln],      bs1 = bias[c0 + 16 + ln];
  float bs2 = bias[c0 + 32 + ln], bs3 = bias[c0 + 48 + ln];
  if (which < 2) {
    _Float16* dst = which ? kb : qb;
    // inv_freq(d) = 10000^(-2d/64) = exp(-d/32 * ln(10000))
    float ifa = expf(-(float)ln * (9.210340371976184f / 32.0f));
    float ifb = expf(-(float)(16 + ln) * (9.210340371976184f / 32.0f));
#pragma unroll
    for (int r = 0; r < 8; ++r) {
      int ts = tsbase + hf * 8 + r;
      _Float16* rowp = dst + (((size_t)(b * NH + h) * TS + ts) << 6);
      float ca = cosf((float)ts * ifa), sa = sinf((float)ts * ifa);
      float x0 = a0[r] + bs0, x1 = a2[r] + bs2;   // pair (d=ln, d+32)
      rowp[ln]      = (_Float16)(x0 * ca - x1 * sa);
      rowp[ln + 32] = (_Float16)(x1 * ca + x0 * sa);
      float cb = cosf((float)ts * ifb), sb = sinf((float)ts * ifb);
      float y0 = a1[r] + bs1, y1 = a3[r] + bs3;   // pair (d=16+ln, d+48)
      rowp[ln + 16] = (_Float16)(y0 * cb - y1 * sb);
      rowp[ln + 48] = (_Float16)(y1 * cb + y0 * sb);
    }
  } else {
#pragma unroll
    for (int r = 0; r < 8; ++r) {
      int ts = tsbase + hf * 8 + r;
      size_t vb = (size_t)(b * NH + h) * HD;
      vT[((vb + ln)      << 10) + ts] = (_Float16)(a0[r] + bs0);
      vT[((vb + 16 + ln) << 10) + ts] = (_Float16)(a1[r] + bs1);
      vT[((vb + 32 + ln) << 10) + ts] = (_Float16)(a2[r] + bs2);
      vT[((vb + 48 + ln) << 10) + ts] = (_Float16)(a3[r] + bs3);
    }
  }
}

// --------------------------- flash attention ---------------------------
// 8 waves in a block share one (b,h); K-panel (32x64, contiguous) and V-panel
// (64x32, strided) are TDM-staged per key chunk, double-buffered.
__global__ void k_attn(const _Float16* __restrict__ qb,
                       const _Float16* __restrict__ kb,
                       const _Float16* __restrict__ vT,
                       _Float16* __restrict__ ao) {
  __shared__ alignas(32) _Float16 pls[8][16 * 32];     // per-wave P staging
  __shared__ alignas(16) _Float16 kpan[2][32 * 64];    // 2 x 4 KB
  __shared__ alignas(16) _Float16 vpan[2][64 * 32];    // 2 x 4 KB
  int wv = threadIdx.x >> 5;
  int w = blockIdx.x * 8 + wv;
  int lane = threadIdx.x & 31;
  int hf = lane >> 4, ln = lane & 15;
  int qtile = w & 63;
  int bh = w >> 6;                         // uniform across the block
  int h = bh & 15, b = bh >> 4;
  int qbase = qtile * 16;
  const size_t bhTS = (size_t)(b * NH + h) * TS;
  const size_t vbase = (size_t)(b * NH + h) * HD;
  const _Float16* qrow = qb + ((bhTS + qbase + ln) << 6);
  v16h a0 = ldA(qrow, 0, hf);              // q is loop-invariant: hoist
  v16h a1 = ldA(qrow, 32, hf);
  v8f o0 = {}, o1 = {}, o2 = {}, o3 = {};
  float mrow[8], lrow[8];
#pragma unroll
  for (int r = 0; r < 8; ++r) { mrow[r] = -1e38f; lrow[r] = 0.f; }
  const float scale = 0.125f;              // 1/sqrt(64)
  _Float16* P = pls[wv];
#ifdef HAVE_TDM
  if (wv == 0) {
    tdm_load(kb + (bhTS << 6), kpan[0], 512, 0, 512);      // 4 KB contiguous
    tdm_load(vT + (vbase << 10), vpan[0], 8, 64, 256);     // 64 rows x 32 f16
  }
#endif
  for (int j = 0; j < TS; j += 32) {
    int cur = (j >> 5) & 1;
#ifdef HAVE_TDM
    if (wv == 0) {
      if (j + 32 < TS) {
        tdm_load(kb + ((bhTS + j + 32) << 6), kpan[cur ^ 1], 512, 0, 512);
        tdm_load(vT + (vbase << 10) + j + 32, vpan[cur ^ 1], 8, 64, 256);
        __builtin_amdgcn_s_wait_tensorcnt(2);   // current pair done
      } else {
        __builtin_amdgcn_s_wait_tensorcnt(0);
      }
    }
#else
    {  // cooperative fallback
      int t = threadIdx.x;
      *(v8h*)(kpan[cur] + t * 8) = *(const v8h*)(kb + ((bhTS + j) << 6) + t * 8);
      *(v8h*)(vpan[cur] + (t >> 2) * 32 + (t & 3) * 8) =
          *(const v8h*)(vT + ((vbase + (t >> 2)) << 10) + j + (t & 3) * 8);
    }
#endif
    __syncthreads();                       // barrier A: panels ready
    const _Float16* kp = kpan[cur];
    const _Float16* vp = vpan[cur];
    v8f s0 = {}, s1 = {};
    s0 = wmma16(a0, ldB(kp + (size_t)(ln)      * 64, 0,  hf), s0);
    s0 = wmma16(a1, ldB(kp + (size_t)(ln)      * 64, 32, hf), s0);
    s1 = wmma16(a0, ldB(kp + (size_t)(16 + ln) * 64, 0,  hf), s1);
    s1 = wmma16(a1, ldB(kp + (size_t)(16 + ln) * 64, 32, hf), s1);
    float alpha[8], psum[8];
#pragma unroll
    for (int r = 0; r < 8; ++r) {
      int qm = qbase + hf * 8 + r;
      float v0 = (j + ln <= qm)      ? s0[r] * scale : -1e38f;
      float v1 = (j + 16 + ln <= qm) ? s1[r] * scale : -1e38f;
      float t = fmaxf(v0, v1);
#pragma unroll
      for (int o = 1; o < 16; o <<= 1) t = fmaxf(t, __shfl_xor(t, o, 32));
      float nm = fmaxf(mrow[r], t);
      alpha[r] = expf(mrow[r] - nm);
      float p0 = expf(v0 - nm), p1 = expf(v1 - nm);
      float ps = p0 + p1;
#pragma unroll
      for (int o = 1; o < 16; o <<= 1) ps += __shfl_xor(ps, o, 32);
      psum[r] = ps;
      mrow[r] = nm;
      int m = hf * 8 + r;
      P[m * 32 + ln]      = (_Float16)p0;  // C-layout -> LDS [row][key]
      P[m * 32 + 16 + ln] = (_Float16)p1;
    }
    __syncthreads();                       // barrier B: P visible
    v16h pa;                               // read back in A-operand layout
    {
      v8h lo = *(const v8h*)(P + ln * 32 + hf * 8);
      v8h hi = *(const v8h*)(P + ln * 32 + 16 + hf * 8);
      pa = make_v16(lo, hi);
    }
#pragma unroll
    for (int r = 0; r < 8; ++r) {
      o0[r] *= alpha[r]; o1[r] *= alpha[r];
      o2[r] *= alpha[r]; o3[r] *= alpha[r];
      lrow[r] = lrow[r] * alpha[r] + psum[r];
    }
    o0 = wmma16(pa, ldB(vp + (size_t)(ln)      * 32, 0, hf), o0);
    o1 = wmma16(pa, ldB(vp + (size_t)(16 + ln) * 32, 0, hf), o1);
    o2 = wmma16(pa, ldB(vp + (size_t)(32 + ln) * 32, 0, hf), o2);
    o3 = wmma16(pa, ldB(vp + (size_t)(48 + ln) * 32, 0, hf), o3);
    __syncthreads();                       // barrier C: reads done before reuse
  }
#pragma unroll
  for (int r = 0; r < 8; ++r) {
    int m = hf * 8 + r;
    float invl = 1.0f / lrow[r];
    size_t rowo = ((size_t)(b * TS + qbase + m) << 10) + h * HD;
    ao[rowo + ln]      = (_Float16)(o0[r] * invl);
    ao[rowo + 16 + ln] = (_Float16)(o1[r] * invl);
    ao[rowo + 32 + ln] = (_Float16)(o2[r] * invl);
    ao[rowo + 48 + ln] = (_Float16)(o3[r] * invl);
  }
}

// --------------------------- output projection ---------------------------
// 8 waves (8 M-tiles) share a 64-column weight panel, TDM-staged and
// double-buffered. Each wave computes 16x64 output.
__global__ void k_oproj(const _Float16* __restrict__ A,
                        const _Float16* __restrict__ WT,
                        const float* __restrict__ bias,
                        float* __restrict__ out) {
  __shared__ alignas(16) _Float16 panel[2][64 * 128];   // 2 x 16 KB
  int wv = threadIdx.x >> 5;
  int lane = threadIdx.x & 31, hf = lane >> 4, ln = lane & 15;
  int mblock = blockIdx.x >> 4;            // 32 blocks of 8 M-tiles
  int npanel = blockIdx.x & 15;            // 16 panels of 64 columns
  int rowbase = (mblock * 8 + wv) * 16;
  int n0 = npanel * 64;
  const _Float16* arow = A + ((size_t)(rowbase + ln) << 10);
  const _Float16* wsrc = WT + ((size_t)n0 << 10);
  v8f acc[4] = {{}, {}, {}, {}};
#ifdef HAVE_TDM
  if (wv == 0) tdm_load(wsrc, panel[0], 32, 64, 256);
#endif
  for (int kc = 0; kc < DD; kc += 128) {
    int cur = (kc >> 7) & 1;
#ifdef HAVE_TDM
    if (wv == 0) {
      if (kc + 128 < DD) {
        tdm_load(wsrc + kc + 128, panel[cur ^ 1], 32, 64, 256);
        __builtin_amdgcn_s_wait_tensorcnt(1);
      } else {
        __builtin_amdgcn_s_wait_tensorcnt(0);
      }
    }
#else
    {  // cooperative fallback: 256 threads x 32 f16
      int t = threadIdx.x;
      const _Float16* src = wsrc + ((size_t)(t >> 2) << 10) + kc + (t & 3) * 32;
      _Float16* dst = panel[cur] + (t >> 2) * 128 + (t & 3) * 32;
#pragma unroll
      for (int u = 0; u < 4; ++u)
        *(v8h*)(dst + u * 8) = *(const v8h*)(src + u * 8);
    }
#endif
    __syncthreads();
    const _Float16* pb = panel[cur];
#pragma unroll
    for (int k0 = 0; k0 < 128; k0 += 32) {
      v16h a = ldA(arow, kc + k0, hf);
#pragma unroll
      for (int t = 0; t < 4; ++t)
        acc[t] = wmma16(a, ldB(pb + (size_t)(t * 16 + ln) * 128, k0, hf), acc[t]);
    }
    __syncthreads();
  }
#pragma unroll
  for (int t = 0; t < 4; ++t) {
    float bv = bias[n0 + t * 16 + ln];
#pragma unroll
    for (int r = 0; r < 8; ++r) {
      int m = hf * 8 + r;
      out[((size_t)(rowbase + m) << 10) + n0 + t * 16 + ln] = acc[t][r] + bv;
    }
  }
}

// --------------------------- upsample + residual ---------------------------
__global__ void k_upsample(const float* __restrict__ x,
                           const float* __restrict__ proj,
                           float* __restrict__ out) {
  size_t i = (size_t)blockIdx.x * blockDim.x + threadIdx.x;  // B*T*D/4
  size_t d4 = i & 255;                     // D/4 = 256
  size_t bt = i >> 8;
  int t = (int)(bt & 4095);
  int b = (int)(bt >> 12);
  float src = ((float)t + 0.5f) * 0.25f - 0.5f;
  src = fminf(fmaxf(src, 0.0f), 1023.0f);
  int i0 = (int)floorf(src);
  int i1 = min(i0 + 1, 1023);
  float wt = src - (float)i0;
  float4 a  = ((const float4*)(proj + (((size_t)b * TS + i0) << 10)))[d4];
  float4 c  = ((const float4*)(proj + (((size_t)b * TS + i1) << 10)))[d4];
  float4 xv = ((const float4*)(x + (((size_t)b * TF + t) << 10)))[d4];
  float4 o;
  o.x = xv.x + a.x * (1.0f - wt) + c.x * wt;
  o.y = xv.y + a.y * (1.0f - wt) + c.y * wt;
  o.z = xv.z + a.z * (1.0f - wt) + c.z * wt;
  o.w = xv.w + a.w * (1.0f - wt) + c.w * wt;
  ((float4*)out)[i] = o;
}

// ---------------------------------------------------------------------------
extern "C" void kernel_launch(void* const* d_in, const int* in_sizes, int n_in,
                              void* d_out, int out_size, void* d_ws,
                              size_t ws_size, hipStream_t stream) {
  const float* x      = (const float*)d_in[0];
  const float* norm_w = (const float*)d_in[1];
  const float* norm_b = (const float*)d_in[2];
  const float* W_qkv  = (const float*)d_in[3];
  const float* b_qkv  = (const float*)d_in[4];
  const float* W_out  = (const float*)d_in[5];
  const float* b_out  = (const float*)d_in[6];
  float* out = (float*)d_out;

  char* ws = (char*)d_ws;
  _Float16* xs    = (_Float16*)(ws);
  _Float16* WqkvT = (_Float16*)(ws + ((size_t)8 << 20));
  _Float16* WoutT = (_Float16*)(ws + ((size_t)14 << 20));
  _Float16* qbuf  = (_Float16*)(ws + ((size_t)16 << 20));
  _Float16* kbuf  = (_Float16*)(ws + ((size_t)24 << 20));
  _Float16* vTbuf = (_Float16*)(ws + ((size_t)32 << 20));
  _Float16* aobuf = (_Float16*)(ws + ((size_t)40 << 20));
  float*    proj  = (float*)   (ws + ((size_t)48 << 20));

  k_t_qkv<<<12288, 256, 0, stream>>>(W_qkv, WqkvT);
  k_t_out<<<4096, 256, 0, stream>>>(W_out, WoutT);
  k_ln<<<4096, 256, 0, stream>>>(x, norm_w, norm_b, xs);
  k_qkv<<<1536, 256, 0, stream>>>(xs, WqkvT, b_qkv, qbuf, kbuf, vTbuf);
  k_attn<<<512, 256, 0, stream>>>(qbuf, kbuf, vTbuf, aobuf);
  k_oproj<<<512, 256, 0, stream>>>(aobuf, WoutT, b_out, proj);
  k_upsample<<<16384, 256, 0, stream>>>(x, proj, out);
}
